// SEIRPhysicsPINN_37898791420423
// MI455X (gfx1250) — matrix operand order repaired and checked by reading, not compiled, not run on hardware
//
#include <hip/hip_runtime.h>
#include <hip/hip_bf16.h>
#include <stdint.h>

// ---------------------------------------------------------------------------
// Types for WMMA (CDNA5 gfx1250, wave32): D(16x16 f32) = A(16x32 bf16) x B(32x16 bf16) + C
// ---------------------------------------------------------------------------
typedef __attribute__((ext_vector_type(16))) __bf16 v16bf;
typedef __attribute__((ext_vector_type(8)))  float  v8f;

union ABfrag { v16bf v; uint32_t u[8]; };

static __device__ __forceinline__ uint32_t bf16rne(float x) {
    uint32_t u = __float_as_uint(x);
    return (u + 0x7FFFu + ((u >> 16) & 1u)) >> 16;   // round-to-nearest-even
}
static __device__ __forceinline__ float bf16tof(uint32_t h) {
    return __uint_as_float(h << 16);
}

// ---------------------------------------------------------------------------
// Elementwise f32 -> packed bf16 convert (pairs). npairs = nelem/2.
// ---------------------------------------------------------------------------
__global__ __launch_bounds__(256) void fconv_pk(const float* __restrict__ src,
                                                uint16_t* __restrict__ dst,
                                                int npairs) {
    int i = blockIdx.x * 256 + threadIdx.x;
    if (i < npairs) {
        float2 f = ((const float2*)src)[i];
        ((uint32_t*)dst)[i] = bf16rne(f.x) | (bf16rne(f.y) << 16);
    }
}

// ---------------------------------------------------------------------------
// Transpose-convert: src f32 (R x C, row-major) -> dst bf16 (C x R, row-major)
// block (32,8), grid (C/32, R/32). All shapes here are multiples of 32.
// ---------------------------------------------------------------------------
__global__ __launch_bounds__(256) void tconv(const float* __restrict__ src,
                                             uint16_t* __restrict__ dst,
                                             int R, int C) {
    __shared__ float t[32][33];
    int lx = threadIdx.x, ly = threadIdx.y;
    int c0 = blockIdx.x * 32, r0 = blockIdx.y * 32;
#pragma unroll
    for (int i = 0; i < 32; i += 8) {
        t[ly + i][lx] = src[(size_t)(r0 + ly + i) * C + (c0 + lx)];
    }
    __syncthreads();
#pragma unroll
    for (int i = 0; i < 32; i += 8) {
        dst[(size_t)(c0 + ly + i) * R + (r0 + lx)] = (uint16_t)bf16rne(t[lx][ly + i]);
    }
}

// ---------------------------------------------------------------------------
// Wave-per-16x16-tile WMMA GEMM:  C(MxN) = epi( A(MxK) * B(KxN) + bias[n] )
//   A: bf16 row-major (M x K).  B: TRANSPOSED bf16 BT (N x K, row-major) so
//   both fragments load as packed 32-bit k-pairs.
// EPI: 0 = relu -> bf16 store;  1 = sigmoid*scale -> f32 store (params);
//      2 = plain -> bf16 store (logits)
// Block = 256 threads = 8 waves; tile counts are exact multiples of 8.
// ---------------------------------------------------------------------------
template<int EPI>
__global__ __launch_bounds__(256) void gemm_wmma_k(
    const uint16_t* __restrict__ A, const uint16_t* __restrict__ Bt,
    const float* __restrict__ bias, void* __restrict__ Cout,
    int M, int N, int K)
{
    int tid  = threadIdx.x;
    int wave = tid >> 5, lane = tid & 31;
    int64_t tile = (int64_t)blockIdx.x * 8 + wave;
    int ntn = N >> 4;
    int mt = (int)(tile / ntn), nt = (int)(tile % ntn);
    if (mt >= (M >> 4)) return;

    int mr = lane & 15, hi = lane >> 4;
    int arow = mt * 16 + mr;   // A matrix row (M index) owned by this lane
    int brow = nt * 16 + mr;   // BT row (= B column / N index) owned by this lane

    v8f acc = {0.f, 0.f, 0.f, 0.f, 0.f, 0.f, 0.f, 0.f};

    for (int kb = 0; kb < K; kb += 32) {
        // speculative prefetch of next K-slab of B (harmlessly dropped at tail)
        __builtin_prefetch(Bt + (size_t)brow * K + kb + 32, 0, 1);
        ABfrag a, b;
        // A 16x32 bf16 layout: lanes 0-15 hold M=lane, K pairs {0..7,16..23};
        // lanes 16-31 hold K pairs {8..15,24..31}.
#pragma unroll
        for (int j = 0; j < 8; ++j) {
            int k = kb + ((j < 4) ? (8 * hi + 2 * j) : (16 + 8 * hi + 2 * (j - 4)));
            a.u[j] = *(const uint32_t*)(A + (size_t)arow * K + k);
        }
        // B 32x16 bf16 layout: same K order as A; lanes 0-15 hold K {0..7,16..23},
        // lanes 16-31 hold K {8..15,24..31} for column n = lane&15.
#pragma unroll
        for (int j = 0; j < 8; ++j) {
            int k = kb + ((j < 4) ? (8 * hi + 2 * j) : (16 + 8 * hi + 2 * (j - 4)));
            b.u[j] = *(const uint32_t*)(Bt + (size_t)brow * K + k);
        }
        acc = __builtin_amdgcn_wmma_f32_16x16x32_bf16(false, a.v, false, b.v,
                                                      (short)0, acc, false, false);
    }

    // C/D layout: lane holds column n = lane&15; VGPR r holds row m = r + 8*hi.
    int n = nt * 16 + (lane & 15);
    float bv = bias[n];
#pragma unroll
    for (int r = 0; r < 8; ++r) {
        int m = mt * 16 + r + 8 * hi;
        float v = acc[r] + bv;
        if (EPI == 0) {
            v = fmaxf(v, 0.f);
            ((uint16_t*)Cout)[(size_t)m * N + n] = (uint16_t)bf16rne(v);
        } else if (EPI == 1) {
            float scale = (n < 256) ? 3.f : ((n < 768) ? 1.f : 5.f);
            ((float*)Cout)[(size_t)m * N + n] = scale / (1.f + __expf(-v));
        } else {
            ((uint16_t*)Cout)[(size_t)m * N + n] = (uint16_t)bf16rne(v);
        }
    }
}

// ---------------------------------------------------------------------------
// In-place row softmax over bf16 logits: one 256-thread block per length-256 row.
// Grid = 512*256 = 131072 blocks.
// ---------------------------------------------------------------------------
__global__ __launch_bounds__(256) void softmax_rows(uint16_t* __restrict__ logits) {
    __shared__ float red[256];
    size_t base = (size_t)blockIdx.x * 256;
    int t = threadIdx.x;
    float x = bf16tof(logits[base + t]);
    red[t] = x;
    __syncthreads();
    for (int s = 128; s > 0; s >>= 1) {
        if (t < s) red[t] = fmaxf(red[t], red[t + s]);
        __syncthreads();
    }
    float mx = red[0];
    __syncthreads();
    float e = __expf(x - mx);
    red[t] = e;
    __syncthreads();
    for (int s = 128; s > 0; s >>= 1) {
        if (t < s) red[t] += red[t + s];
        __syncthreads();
    }
    float inv = 1.f / red[0];
    logits[base + t] = (uint16_t)bf16rne(e * inv);
}

// ---------------------------------------------------------------------------
// SEIR RK4 simulation: one workgroup (256 threads = 8 wave32) per batch element.
// Pi[b] (256x256 bf16 = 128 KB) staged in dynamic LDS (2 WGs / 320KB WGP).
// Matvec force = I . Pi: thread t = (half, m2) computes columns {2m2, 2m2+1}
// over K-range half*128..+128 with ONE packed ds_load_b32 per n (two bf16
// columns per dword, consecutive lanes -> consecutive banks, conflict-free),
// then a 512-float LDS exchange combines the two K-halves.
// ---------------------------------------------------------------------------
__device__ __forceinline__ float seir_force(const uint32_t* __restrict__ pil32,
                                            float* __restrict__ I_sh,
                                            float* __restrict__ fbuf,
                                            int m, int half, int m2, float Iv) {
    __syncthreads();          // previous-stage reads of I_sh / fbuf complete
    I_sh[m] = Iv;
    __syncthreads();
    float fe = 0.f, fo = 0.f;
    int nbase = half << 7;
#pragma unroll 4
    for (int nn = 0; nn < 128; ++nn) {
        int n = nbase + nn;
        uint32_t p = pil32[(n << 7) + m2];       // bf16 cols {2m2, 2m2+1} of row n
        float iv = I_sh[n];                      // broadcast (vectorizes to b128)
        fe = fmaf(iv, __uint_as_float(p << 16), fe);
        fo = fmaf(iv, __uint_as_float(p & 0xFFFF0000u), fo);
    }
    fbuf[(half << 8) + (m2 << 1)]     = fe;
    fbuf[(half << 8) + (m2 << 1) + 1] = fo;
    __syncthreads();
    return fbuf[m] + fbuf[256 + m];
}

__global__ __launch_bounds__(256) void seir_sim(const uint16_t* __restrict__ Pi,
                                                const float* __restrict__ par,
                                                const float* __restrict__ x_hist,
                                                float* __restrict__ out) {
    extern __shared__ uint16_t pil[];     // 65536 x bf16 = 128 KB, row-major
    __shared__ float I_sh[256];
    __shared__ float fbuf[512];
    int b = blockIdx.x, m = threadIdx.x;
    int half = m >> 7, m2 = m & 127;

    {   // stage Pi[b] into LDS with 128-bit copies
        const uint4* src = (const uint4*)(Pi + (size_t)b * 65536);
        uint4* dst = (uint4*)pil;
        for (int i = m; i < 8192; i += 256) dst[i] = src[i];
    }
    const uint32_t* pil32 = (const uint32_t*)pil;

    size_t pb = (size_t)b * 1024;
    float beta = par[pb + m];
    float sg   = par[pb + 256 + m];
    float gm   = par[pb + 512 + m];
    float e0r  = par[pb + 768 + m];

    float I = fmaxf(x_hist[(size_t)b * 4096 + 15 * 256 + m], 1e-6f);
    float E = I * e0r;
    float S = fmaxf(1.f - E - I, 0.01f);

    const float DT = 0.25f;
    for (int week = 0; week < 12; ++week) {
        for (int st = 0; st < 4; ++st) {
            float f1 = seir_force(pil32, I_sh, fbuf, m, half, m2, I);
            float ni1 = beta * S * f1;
            float dS1 = -ni1, dE1 = ni1 - sg * E, dI1 = sg * E - gm * I;

            float S2 = S + 0.5f * DT * dS1, E2 = E + 0.5f * DT * dE1, I2 = I + 0.5f * DT * dI1;
            float f2 = seir_force(pil32, I_sh, fbuf, m, half, m2, I2);
            float ni2 = beta * S2 * f2;
            float dS2 = -ni2, dE2 = ni2 - sg * E2, dI2 = sg * E2 - gm * I2;

            float S3 = S + 0.5f * DT * dS2, E3 = E + 0.5f * DT * dE2, I3 = I + 0.5f * DT * dI2;
            float f3 = seir_force(pil32, I_sh, fbuf, m, half, m2, I3);
            float ni3 = beta * S3 * f3;
            float dS3 = -ni3, dE3 = ni3 - sg * E3, dI3 = sg * E3 - gm * I3;

            float S4 = S + DT * dS3, E4 = E + DT * dE3, I4 = I + DT * dI3;
            float f4 = seir_force(pil32, I_sh, fbuf, m, half, m2, I4);
            float ni4 = beta * S4 * f4;
            float dS4 = -ni4, dE4 = ni4 - sg * E4, dI4 = sg * E4 - gm * I4;

            S = fminf(fmaxf(S + DT / 6.f * (dS1 + 2.f * dS2 + 2.f * dS3 + dS4), 0.f), 1.f);
            E = fminf(fmaxf(E + DT / 6.f * (dE1 + 2.f * dE2 + 2.f * dE3 + dE4), 0.f), 1.f);
            I = fminf(fmaxf(I + DT / 6.f * (dI1 + 2.f * dI2 + 2.f * dI3 + dI4), 0.f), 1.f);
        }
        out[(size_t)b * 3072 + week * 256 + m]           = I;   // I_sim
        out[1572864 + (size_t)b * 3072 + week * 256 + m] = E;   // E_sim
    }
}

// ---------------------------------------------------------------------------
// Reductions for the mean outputs
// ---------------------------------------------------------------------------
__global__ __launch_bounds__(256) void param_means(const float* __restrict__ par,
                                                   float* __restrict__ out) {
    int j = blockIdx.x * 256 + threadIdx.x;   // 0..767 -> beta|sigma|gamma cols
    float s = 0.f;
    for (int b = 0; b < 512; ++b) s += par[(size_t)b * 1024 + j];
    out[3145728 + j] = s * (1.f / 512.f);
}

__global__ __launch_bounds__(256) void pi_mean(const uint16_t* __restrict__ Pi,
                                               float* __restrict__ out) {
    int p = blockIdx.x * 256 + threadIdx.x;   // 0..65535
    float s = 0.f;
    for (int b = 0; b < 512; ++b) s += bf16tof(Pi[(size_t)b * 65536 + p]);
    out[3146496 + p] = s * (1.f / 512.f);
}

// ---------------------------------------------------------------------------
// Launch
// ---------------------------------------------------------------------------
extern "C" void kernel_launch(void* const* d_in, const int* in_sizes, int n_in,
                              void* d_out, int out_size, void* d_ws, size_t ws_size,
                              hipStream_t stream) {
    (void)in_sizes; (void)n_in; (void)out_size; (void)ws_size;
    const float* x_hist = (const float*)d_in[0];   // (512,16,256)
    const float* cnn    = (const float*)d_in[1];   // (512,64)
    const float* W1     = (const float*)d_in[2];   // (4096,128)
    const float* b1     = (const float*)d_in[3];   // (128)
    const float* W2     = (const float*)d_in[4];   // (128,1024)
    const float* b2     = (const float*)d_in[5];   // (1024)
    const float* M1     = (const float*)d_in[6];   // (64,128)
    const float* mb1    = (const float*)d_in[7];   // (128)
    const float* M2     = (const float*)d_in[8];   // (128,65536)
    const float* mb2    = (const float*)d_in[9];   // (65536)
    float* out = (float*)d_out;

    char* ws = (char*)d_ws;
    size_t off = 0;
    auto alloc = [&](size_t bytes) -> char* {
        char* p = ws + off;
        off += (bytes + 255) & ~(size_t)255;
        return p;
    };
    uint16_t* Xb  = (uint16_t*)alloc(512ull * 4096 * 2);   // x_hist bf16
    uint16_t* Cb  = (uint16_t*)alloc(512ull * 64 * 2);     // cnn_feat bf16
    uint16_t* W1T = (uint16_t*)alloc(128ull * 4096 * 2);   // W1^T bf16 (128 x 4096)
    uint16_t* W2T = (uint16_t*)alloc(1024ull * 128 * 2);   // W2^T bf16 (1024 x 128)
    uint16_t* M1T = (uint16_t*)alloc(128ull * 64 * 2);     // M1^T bf16 (128 x 64)
    uint16_t* M2T = (uint16_t*)alloc(65536ull * 128 * 2);  // M2^T bf16 (65536 x 128)
    uint16_t* hb  = (uint16_t*)alloc(512ull * 128 * 2);    // relu(X@W1+b1) bf16
    uint16_t* hmb = (uint16_t*)alloc(512ull * 128 * 2);    // relu(cnn@M1+mb1) bf16
    float*    par = (float*)   alloc(512ull * 1024 * 4);   // scaled sigmoids f32
    uint16_t* Pi  = (uint16_t*)alloc(512ull * 65536 * 2);  // logits -> softmax in place

    // operand conversions (one-time, coalesced)
    fconv_pk<<<(512 * 4096 / 2) / 256, 256, 0, stream>>>(x_hist, Xb, 512 * 4096 / 2);
    fconv_pk<<<(512 * 64 / 2) / 256, 256, 0, stream>>>(cnn, Cb, 512 * 64 / 2);
    dim3 tb(32, 8);
    tconv<<<dim3(128 / 32, 4096 / 32), tb, 0, stream>>>(W1, W1T, 4096, 128);
    tconv<<<dim3(1024 / 32, 128 / 32), tb, 0, stream>>>(W2, W2T, 128, 1024);
    tconv<<<dim3(128 / 32, 64 / 32),   tb, 0, stream>>>(M1, M1T, 64, 128);
    tconv<<<dim3(65536 / 32, 128 / 32), tb, 0, stream>>>(M2, M2T, 128, 65536);

    // h = relu(X @ W1 + b1): 512x128, K=4096 -> 256 tiles / 8 waves
    gemm_wmma_k<0><<<32, 256, 0, stream>>>(Xb, W1T, b1, hb, 512, 128, 4096);
    // params = sigmoid-scale(h @ W2 + b2): 512x1024 -> 2048 tiles
    gemm_wmma_k<1><<<256, 256, 0, stream>>>(hb, W2T, b2, par, 512, 1024, 128);
    // hm = relu(cnn @ M1 + mb1): 512x128, K=64 -> 256 tiles
    gemm_wmma_k<0><<<32, 256, 0, stream>>>(Cb, M1T, mb1, hmb, 512, 128, 64);
    // logits = hm @ M2 + mb2: 512x65536 -> 131072 tiles
    gemm_wmma_k<2><<<16384, 256, 0, stream>>>(hmb, M2T, mb2, Pi, 512, 65536, 128);

    softmax_rows<<<131072, 256, 0, stream>>>(Pi);

    seir_sim<<<512, 256, 65536 * sizeof(uint16_t), stream>>>(Pi, par, x_hist, out);

    param_means<<<3, 256, 0, stream>>>(par, out);
    pi_mean<<<256, 256, 0, stream>>>(Pi, out);
}
